// Net_4337916969750
// MI455X (gfx1250) — matrix operand order repaired and checked by reading, not compiled, not run on hardware
//
#include <hip/hip_runtime.h>

// ---------------------------------------------------------------------------
// Types
// ---------------------------------------------------------------------------
typedef __bf16 bf16_t;
typedef __attribute__((ext_vector_type(16))) __bf16 bf16x16;
typedef __attribute__((ext_vector_type(8)))  float  floatx8;
typedef __attribute__((ext_vector_type(4)))  unsigned int uintx4;

#define EPS 1e-5f

union bfcast {
  uintx4 u4[2];
  bf16x16 v;
};

// ---------------------------------------------------------------------------
// WMMA fragment helpers (layouts per CDNA5 ISA §7.12.2, wave32)
// ---------------------------------------------------------------------------
// A fragment: 16x32 bf16 from row-major source. Row = lane&15; K chunks at
// half*8 (16B) and 16+half*8 (16B), half = lane>>4. Two b128 loads per lane.
__device__ __forceinline__ bf16x16 load_a_frag(const bf16_t* A, int lda) {
  int lane = threadIdx.x & 31;
  const bf16_t* p = A + (size_t)(lane & 15) * lda + ((lane >> 4) * 8);
  bfcast c;
  c.u4[0] = *reinterpret_cast<const uintx4*>(p);
  c.u4[1] = *reinterpret_cast<const uintx4*>(p + 16);
  return c.v;
}

// B fragment: 32x16 bf16 with B[k][n] = W[n*ldw + k] (Linear W[N,K] row-major).
// Lane holds column n = n0+(lane&15), K chunks at k0+(lane>>4)*8 (16B) and
// k0+16+(lane>>4)*8 (16B) -> two b128 loads per lane.
__device__ __forceinline__ bf16x16 load_b_frag_w(const bf16_t* __restrict__ W,
                                                 int ldw, int n0, int k0) {
  int lane = threadIdx.x & 31;
  const bf16_t* p = W + (size_t)(n0 + (lane & 15)) * ldw + k0 + ((lane >> 4) * 8);
  bfcast c;
  c.u4[0] = *reinterpret_cast<const uintx4*>(p);
  c.u4[1] = *reinterpret_cast<const uintx4*>(p + 16);
  return c.v;
}

// ---------------------------------------------------------------------------
// conv1 (7x7 s2 p3) + BN + ReLU, fp32 direct, bf16 out. [32,3,224,224]->[32,64,112,112]
// ---------------------------------------------------------------------------
__global__ void conv1_bn_relu_kernel(const float* __restrict__ in,
                                     const float* __restrict__ w,
                                     const float* __restrict__ g,
                                     const float* __restrict__ bb,
                                     const float* __restrict__ mm,
                                     const float* __restrict__ vv,
                                     bf16_t* __restrict__ out, long total) {
  long o = (long)blockIdx.x * blockDim.x + threadIdx.x;
  if (o >= total) return;
  int q = (int)(o % 112); long t1 = o / 112;
  int p = (int)(t1 % 112); long t2 = t1 / 112;
  int k = (int)(t2 % 64);  int b = (int)(t2 / 64);
  const float* wk  = w + (size_t)k * 3 * 49;
  const float* inb = in + (size_t)b * 3 * 224 * 224;
  int ih0 = p * 2 - 3, iw0 = q * 2 - 3;
  float sum = 0.f;
  for (int c = 0; c < 3; ++c)
    for (int r = 0; r < 7; ++r) {
      int ih = ih0 + r;
      if ((unsigned)ih >= 224u) continue;
      const float* row = inb + ((size_t)c * 224 + ih) * 224;
      const float* wr  = wk + (c * 7 + r) * 7;
      for (int s = 0; s < 7; ++s) {
        int iw = iw0 + s;
        if ((unsigned)iw >= 224u) continue;
        sum += row[iw] * wr[s];
      }
    }
  float inv = g[k] * rsqrtf(vv[k] + EPS);
  float val = fmaxf(sum * inv + (bb[k] - mm[k] * inv), 0.f);
  out[o] = (bf16_t)val;
}

// maxpool 3x3 s2 p1: [32,64,112,112] bf16 -> [32,64,56,56] bf16
__global__ void maxpool_kernel(const bf16_t* __restrict__ in,
                               bf16_t* __restrict__ out, int total) {
  int o = blockIdx.x * blockDim.x + threadIdx.x;
  if (o >= total) return;
  int q = o % 56; int t1 = o / 56;
  int p = t1 % 56; int bc = t1 / 56;
  const bf16_t* src = in + (size_t)bc * 112 * 112;
  int ih0 = p * 2 - 1, iw0 = q * 2 - 1;
  float best = -3.4e38f;
  for (int r = 0; r < 3; ++r) {
    int ih = ih0 + r; if ((unsigned)ih >= 112u) continue;
    for (int s = 0; s < 3; ++s) {
      int iw = iw0 + s; if ((unsigned)iw >= 112u) continue;
      best = fmaxf(best, (float)src[ih * 112 + iw]);
    }
  }
  out[o] = (bf16_t)best;
}

// ---------------------------------------------------------------------------
// Implicit-GEMM WMMA conv 3x3 (pad 1) + BN + ReLU, fully templated geometry so
// all index math is constant-divide (mul-shift), no runtime division.
// GEMM: out[m,n] = sum_k im2col[m,k] * W[n,k], m = b*P*Q + p*Q + q, k = c*9+r*3+s.
// Block: 256 thr (8 waves = 2 M-waves x 4 N-waves), tile 64(M) x 128(N).
// A tile (64x32 bf16) staged in LDS per k-step; B frags b128-loaded from weights.
// OUT_MODE 0: bf16 NCHW; OUT_MODE 1: fp32 attention-memory layout [B,196,COUT].
// ---------------------------------------------------------------------------
template <int CIN, int HIN, int WIN, int COUT, int P, int Q, int STRIDE, int OUT_MODE>
__global__ void __launch_bounds__(256)
conv_wmma_kernel(const bf16_t* __restrict__ in, const bf16_t* __restrict__ W,
                 const float* __restrict__ g, const float* __restrict__ bb,
                 const float* __restrict__ mm, const float* __restrict__ vv,
                 void* __restrict__ out) {
  constexpr int Kdim = CIN * 9;
  constexpr int PQ = P * Q;
  __shared__ bf16_t Atile[64 * 40];  // stride 40 to spread banks

  const int blockM0 = blockIdx.x * 64;
  const int blockN0 = blockIdx.y * 128;
  const int tid = threadIdx.x;
  const int lane = tid & 31;
  const int wave = tid >> 5;
  const int wave_m = wave & 1;   // rows [wave_m*32, +32)
  const int wave_n = wave >> 1;  // cols [wave_n*32, +32)

  // Per-thread im2col staging coordinates: row = tid>>2, k-segment = (tid&3)*8
  const int arow = tid >> 2;
  const int kseg = (tid & 3) * 8;
  const int m = blockM0 + arow;
  const int bimg = m / PQ;           // constant divide
  const int pq = m - bimg * PQ;
  const int p = pq / Q;              // constant divide
  const int q = pq - p * Q;
  const int ih0 = p * STRIDE - 1;
  const int iw0 = q * STRIDE - 1;
  const bf16_t* inB = in + (size_t)bimg * CIN * HIN * WIN;

  floatx8 acc[2][2];
#pragma unroll
  for (int i = 0; i < 2; ++i)
#pragma unroll
    for (int j = 0; j < 2; ++j)
#pragma unroll
      for (int e = 0; e < 8; ++e) acc[i][j][e] = 0.f;

  for (int k0 = 0; k0 < Kdim; k0 += 32) {
    __syncthreads();
#pragma unroll
    for (int i = 0; i < 8; ++i) {
      int kk = k0 + kseg + i;
      int c = kk / 9;                // constant divide
      int rs = kk - c * 9;
      int r = rs / 3;                // constant divide
      int s = rs - r * 3;
      int ih = ih0 + r;
      int iw = iw0 + s;
      bf16_t val = (bf16_t)0.0f;
      if ((unsigned)ih < (unsigned)HIN && (unsigned)iw < (unsigned)WIN)
        val = inB[((size_t)c * HIN + ih) * WIN + iw];
      Atile[arow * 40 + kseg + i] = val;
    }
    __syncthreads();

    bf16x16 a0 = load_a_frag(&Atile[(wave_m * 32) * 40], 40);
    bf16x16 a1 = load_a_frag(&Atile[(wave_m * 32 + 16) * 40], 40);
#pragma unroll
    for (int j = 0; j < 2; ++j) {
      int n0 = blockN0 + wave_n * 32 + j * 16;
      bf16x16 bf = load_b_frag_w(W, Kdim, n0, k0);
      acc[0][j] = __builtin_amdgcn_wmma_f32_16x16x32_bf16(
          false, a0, false, bf, (short)0, acc[0][j], false, false);
      acc[1][j] = __builtin_amdgcn_wmma_f32_16x16x32_bf16(
          false, a1, false, bf, (short)0, acc[1][j], false, false);
    }
  }

  // Epilogue: BN + ReLU, scatter to NCHW (bf16) or memory layout (fp32)
  const int half = lane >> 4;
  const int ncol = lane & 15;
#pragma unroll
  for (int im = 0; im < 2; ++im) {
    int mbase = blockM0 + wave_m * 32 + im * 16 + half * 8;
#pragma unroll
    for (int j = 0; j < 2; ++j) {
      int n = blockN0 + wave_n * 32 + j * 16 + ncol;
      float inv  = g[n] * rsqrtf(vv[n] + EPS);
      float bias = bb[n] - mm[n] * inv;
#pragma unroll
      for (int r = 0; r < 8; ++r) {
        int mg = mbase + r;
        int bi = mg / PQ;            // constant divide
        int pqi = mg - bi * PQ;
        float val = fmaxf(acc[im][j][r] * inv + bias, 0.f);
        if (OUT_MODE == 0) {
          ((bf16_t*)out)[((size_t)bi * COUT + n) * PQ + pqi] = (bf16_t)val;
        } else if (pqi < 196) {
          ((float*)out)[((size_t)bi * 196 + pqi) * COUT + n] = val;
        }
      }
    }
  }
}

// ---------------------------------------------------------------------------
// Generic WMMA Linear: C[M,N] = A[M,K](bf16) @ W[N,K]^T(bf16) (+bias) (+=C)
// One wave per 16 x (NFRAGS*16) tile; A-frag reused across all N-frags.
// ---------------------------------------------------------------------------
template <int NFRAGS>
__global__ void __launch_bounds__(128)
linear_wmma_kernel(const bf16_t* __restrict__ A, int lda,
                   const bf16_t* __restrict__ W, int ldw,
                   float* __restrict__ C, long ldc,
                   const float* __restrict__ bias,
                   int M, int N, int K, int accumulate, int total_waves) {
  int wave = blockIdx.x * (blockDim.x >> 5) + (threadIdx.x >> 5);
  if (wave >= total_waves) return;  // wave-uniform: EXEC stays all-ones for WMMA

  const int tilesM = M >> 4;
  const int mi = wave % tilesM;
  const int n0 = (wave / tilesM) * (NFRAGS * 16);
  const int lane = threadIdx.x & 31;

  floatx8 acc[NFRAGS];
#pragma unroll
  for (int j = 0; j < NFRAGS; ++j)
#pragma unroll
    for (int e = 0; e < 8; ++e) acc[j][e] = 0.f;

  const bf16_t* Abase = A + (size_t)(mi * 16) * lda;
  for (int k0 = 0; k0 < K; k0 += 32) {
    bf16x16 a = load_a_frag(Abase + k0, lda);
    __builtin_prefetch(W + (size_t)(n0 + (lane & 15)) * ldw + k0 + 32, 0, 1);
#pragma unroll
    for (int j = 0; j < NFRAGS; ++j) {
      bf16x16 b = load_b_frag_w(W, ldw, n0 + j * 16, k0);
      acc[j] = __builtin_amdgcn_wmma_f32_16x16x32_bf16(
          false, a, false, b, (short)0, acc[j], false, false);
    }
  }

  const int half = lane >> 4;
  const int ncol = lane & 15;
#pragma unroll
  for (int j = 0; j < NFRAGS; ++j) {
    int n = n0 + j * 16 + ncol;
    float bv = bias ? bias[n] : 0.f;
#pragma unroll
    for (int r = 0; r < 8; ++r) {
      int mrow = mi * 16 + half * 8 + r;
      size_t idx = (size_t)mrow * ldc + n;
      float val = acc[j][r] + bv;
      if (accumulate) val += C[idx];
      C[idx] = val;
    }
  }
}

// ---------------------------------------------------------------------------
// Attention: per-batch block. scores = q . memory[b,m,:], softmax, ctx.
// ---------------------------------------------------------------------------
__global__ void __launch_bounds__(256)
attention_kernel(const float* __restrict__ q, const float* __restrict__ memory,
                 float* __restrict__ ctx) {
  int b = blockIdx.x;
  int tid = threadIdx.x;
  __shared__ float sc[256];
  __shared__ float red[256];
  const float* qb  = q + b * 768;
  const float* mem = memory + (size_t)b * 196 * 768;

  float score = -3.4e38f;
  if (tid < 196) {
    const float* mrow = mem + (size_t)tid * 768;
    float s = 0.f;
    for (int h = 0; h < 768; ++h) s += qb[h] * mrow[h];
    score = s;
  }
  sc[tid] = score;
  red[tid] = score;
  __syncthreads();
  for (int off = 128; off > 0; off >>= 1) {
    if (tid < off) red[tid] = fmaxf(red[tid], red[tid + off]);
    __syncthreads();
  }
  float mx = red[0];
  __syncthreads();
  float e = (tid < 196) ? __expf(sc[tid] - mx) : 0.f;
  sc[tid] = e;
  red[tid] = e;
  __syncthreads();
  for (int off = 128; off > 0; off >>= 1) {
    if (tid < off) red[tid] += red[tid + off];
    __syncthreads();
  }
  float inv = 1.f / red[0];
  for (int h = tid; h < 768; h += 256) {
    float s = 0.f;
    for (int m2 = 0; m2 < 196; ++m2) s += sc[m2] * mem[(size_t)m2 * 768 + h];
    ctx[b * 768 + h] = s * inv;
  }
}

// ---------------------------------------------------------------------------
// Small elementwise kernels
// ---------------------------------------------------------------------------
__global__ void f32_to_bf16_kernel(const float* __restrict__ src,
                                   bf16_t* __restrict__ dst, int n) {
  int i = blockIdx.x * blockDim.x + threadIdx.x;
  if (i < n) dst[i] = (bf16_t)src[i];
}

__global__ void embed_kernel(const int* __restrict__ captions,
                             const float* __restrict__ table,
                             bf16_t* __restrict__ out) {
  int idx = blockIdx.x * blockDim.x + threadIdx.x;  // 32*16*768
  if (idx >= 32 * 16 * 768) return;
  int h = idx % 768;
  int bt = idx / 768;
  int tok = captions[bt];
  out[idx] = (bf16_t)table[(size_t)tok * 768 + h];
}

__global__ void init_state_kernel(float* h, float* c, bf16_t* h_bf) {
  int i = blockIdx.x * blockDim.x + threadIdx.x;
  if (i < 32 * 768) { h[i] = 0.f; c[i] = 0.f; h_bf[i] = (bf16_t)0.f; }
}

__global__ void build_xin_kernel(const bf16_t* __restrict__ emb,
                                 const float* __restrict__ ctx,
                                 bf16_t* __restrict__ xin, int t) {
  int idx = blockIdx.x * blockDim.x + threadIdx.x;  // 32*1536
  if (idx >= 32 * 1536) return;
  int b = idx / 1536, h = idx % 1536;
  xin[idx] = (h < 768) ? emb[((size_t)b * 16 + t) * 768 + h]
                       : (bf16_t)ctx[b * 768 + (h - 768)];
}

__global__ void lstm_pointwise_kernel(const float* __restrict__ gates,
                                      float* __restrict__ c,
                                      float* __restrict__ h,
                                      bf16_t* __restrict__ h_bf) {
  int idx = blockIdx.x * blockDim.x + threadIdx.x;  // 32*768
  if (idx >= 32 * 768) return;
  int b = idx / 768, hh = idx % 768;
  const float* gb = gates + (size_t)b * 3072;
  float ig = gb[hh], fg = gb[768 + hh], gg = gb[1536 + hh], og = gb[2304 + hh];
  float si = 1.f / (1.f + __expf(-ig));
  float sf = 1.f / (1.f + __expf(-fg));
  float so = 1.f / (1.f + __expf(-og));
  float cn = sf * c[idx] + si * tanhf(gg);
  float hn = so * tanhf(cn);
  c[idx] = cn;
  h[idx] = hn;
  h_bf[idx] = (bf16_t)hn;
}

__global__ void copy_state_kernel(const float* __restrict__ h,
                                  const float* __restrict__ c,
                                  float* __restrict__ out) {
  int i = blockIdx.x * blockDim.x + threadIdx.x;
  if (i < 24576) out[i] = h[i];
  else if (i < 49152) out[i] = c[i - 24576];
}

// ---------------------------------------------------------------------------
// Host orchestration
// ---------------------------------------------------------------------------
extern "C" void kernel_launch(void* const* d_in, const int* in_sizes, int n_in,
                              void* d_out, int out_size, void* d_ws, size_t ws_size,
                              hipStream_t stream) {
  (void)in_sizes; (void)n_in; (void)out_size; (void)ws_size;
  const float* images   = (const float*)d_in[0];
  const int*   captions = (const int*)d_in[1];
  const float* conv1_w  = (const float*)d_in[2];
  const float* bn1_g = (const float*)d_in[3], *bn1_b = (const float*)d_in[4];
  const float* bn1_m = (const float*)d_in[5], *bn1_v = (const float*)d_in[6];
  const float* conv2_w = (const float*)d_in[7];
  const float* bn2_g = (const float*)d_in[8],  *bn2_b = (const float*)d_in[9];
  const float* bn2_m = (const float*)d_in[10], *bn2_v = (const float*)d_in[11];
  const float* conv3_w = (const float*)d_in[12];
  const float* bn3_g = (const float*)d_in[13], *bn3_b = (const float*)d_in[14];
  const float* bn3_m = (const float*)d_in[15], *bn3_v = (const float*)d_in[16];
  const float* conv4_w = (const float*)d_in[17];
  const float* bn4_g = (const float*)d_in[18], *bn4_b = (const float*)d_in[19];
  const float* bn4_m = (const float*)d_in[20], *bn4_v = (const float*)d_in[21];
  const float* emb_table = (const float*)d_in[22];
  const float* attn_w = (const float*)d_in[23];
  const float* w_ih = (const float*)d_in[24];
  const float* w_hh = (const float*)d_in[25];
  const float* b_ih = (const float*)d_in[26];
  const float* b_hh = (const float*)d_in[27];
  const float* fc_w = (const float*)d_in[28];
  const float* fc_b = (const float*)d_in[29];

  char* ws = (char*)d_ws;
  size_t off = 0;
  auto alloc = [&](size_t bytes) -> void* {
    void* p = ws + off;
    off = (off + bytes + 255) & ~(size_t)255;
    return p;
  };

  bf16_t* x1_bf    = (bf16_t*)alloc((size_t)25690112 * 2);  // [32,64,112,112]
  bf16_t* pool_bf  = (bf16_t*)alloc((size_t)6422528 * 2);   // [32,64,56,56]
  bf16_t* x2_bf    = (bf16_t*)alloc((size_t)12845056 * 2);  // [32,128,56,56]
  bf16_t* x3_bf    = (bf16_t*)alloc((size_t)6422528 * 2);   // [32,256,28,28]
  float*  memory_f = (float*) alloc((size_t)4816896 * 4);   // [32,196,768]
  bf16_t* w2_bf    = (bf16_t*)alloc((size_t)73728 * 2);
  bf16_t* w3_bf    = (bf16_t*)alloc((size_t)294912 * 2);
  bf16_t* w4_bf    = (bf16_t*)alloc((size_t)1769472 * 2);
  bf16_t* attnw_bf = (bf16_t*)alloc((size_t)589824 * 2);
  bf16_t* wih_bf   = (bf16_t*)alloc((size_t)4718592 * 2);
  bf16_t* whh_bf   = (bf16_t*)alloc((size_t)2359296 * 2);
  bf16_t* fcw_bf   = (bf16_t*)alloc((size_t)24576000 * 2);
  bf16_t* emb_bf   = (bf16_t*)alloc((size_t)393216 * 2);    // [32,16,768]
  float*  h_f      = (float*) alloc(24576 * 4);
  float*  c_f      = (float*) alloc(24576 * 4);
  bf16_t* h_bf     = (bf16_t*)alloc(24576 * 2);
  float*  q_f      = (float*) alloc(24576 * 4);
  float*  ctx_f    = (float*) alloc(24576 * 4);
  bf16_t* xin_bf   = (bf16_t*)alloc(49152 * 2);
  float*  gates_f  = (float*) alloc(98304 * 4);

  auto cvt = [&](const float* s, bf16_t* d, int n) {
    f32_to_bf16_kernel<<<(n + 255) / 256, 256, 0, stream>>>(s, d, n);
  };
  cvt(conv2_w, w2_bf, 73728);
  cvt(conv3_w, w3_bf, 294912);
  cvt(conv4_w, w4_bf, 1769472);
  cvt(attn_w, attnw_bf, 589824);
  cvt(w_ih, wih_bf, 4718592);
  cvt(w_hh, whh_bf, 2359296);
  cvt(fc_w, fcw_bf, 24576000);

  // ---- Encoder ----
  {
    long tot = 25690112;
    conv1_bn_relu_kernel<<<(tot + 255) / 256, 256, 0, stream>>>(
        images, conv1_w, bn1_g, bn1_b, bn1_m, bn1_v, x1_bf, tot);
  }
  maxpool_kernel<<<(6422528 + 255) / 256, 256, 0, stream>>>(x1_bf, pool_bf, 6422528);

  // conv2: 64->128, 56x56, s1 p1. M=32*56*56=100352
  conv_wmma_kernel<64, 56, 56, 128, 56, 56, 1, 0>
      <<<dim3(100352 / 64, 1), 256, 0, stream>>>(
          pool_bf, w2_bf, bn2_g, bn2_b, bn2_m, bn2_v, (void*)x2_bf);
  // conv3: 128->256, 56x56 -> 28x28, s2 p1. M=32*28*28=25088
  conv_wmma_kernel<128, 56, 56, 256, 28, 28, 2, 0>
      <<<dim3(25088 / 64, 2), 256, 0, stream>>>(
          x2_bf, w3_bf, bn3_g, bn3_b, bn3_m, bn3_v, (void*)x3_bf);
  // conv4: 256->768, 28x28, s1 p1 -> memory [32,196,768] fp32
  conv_wmma_kernel<256, 28, 28, 768, 28, 28, 1, 1>
      <<<dim3(25088 / 64, 6), 256, 0, stream>>>(
          x3_bf, w4_bf, bn4_g, bn4_b, bn4_m, bn4_v, (void*)memory_f);

  // ---- Decoder setup ----
  embed_kernel<<<(393216 + 255) / 256, 256, 0, stream>>>(captions, emb_table, emb_bf);
  init_state_kernel<<<(24576 + 255) / 256, 256, 0, stream>>>(h_f, c_f, h_bf);

  // NFRAGS=4 (16x64 tile) for small-N GEMMs; NFRAGS=8 (16x128 tile) for wide ones.
  auto linear4 = [&](const bf16_t* A, int lda, const bf16_t* Wm, int ldw,
                     float* C, long ldc, const float* bias,
                     int M, int N, int K, int accum) {
    int waves = (M / 16) * (N / 64);
    linear_wmma_kernel<4><<<(waves + 3) / 4, 128, 0, stream>>>(
        A, lda, Wm, ldw, C, ldc, bias, M, N, K, accum, waves);
  };
  auto linear8 = [&](const bf16_t* A, int lda, const bf16_t* Wm, int ldw,
                     float* C, long ldc, const float* bias,
                     int M, int N, int K, int accum) {
    int waves = (M / 16) * (N / 128);
    linear_wmma_kernel<8><<<(waves + 3) / 4, 128, 0, stream>>>(
        A, lda, Wm, ldw, C, ldc, bias, M, N, K, accum, waves);
  };

  float* out = (float*)d_out;
  for (int t = 0; t < 16; ++t) {
    // q = h @ attn_w.T        [32,768]
    linear4(h_bf, 768, attnw_bf, 768, q_f, 768, nullptr, 32, 768, 768, 0);
    // attention -> ctx        [32,768]
    attention_kernel<<<32, 256, 0, stream>>>(q_f, memory_f, ctx_f);
    // xin = concat(e_t, ctx)  [32,1536] bf16
    build_xin_kernel<<<(49152 + 255) / 256, 256, 0, stream>>>(emb_bf, ctx_f, xin_bf, t);
    // gates = xin@w_ih.T + b_ih + h@w_hh.T + b_hh
    linear8(xin_bf, 1536, wih_bf, 1536, gates_f, 3072, b_ih, 32, 3072, 1536, 0);
    linear8(h_bf, 768, whh_bf, 768, gates_f, 3072, b_hh, 32, 3072, 768, 1);
    // LSTM pointwise -> c, h, h_bf
    lstm_pointwise_kernel<<<(24576 + 255) / 256, 256, 0, stream>>>(gates_f, c_f, h_f, h_bf);
    // logits[:, t, :] = h_new @ fc_w.T + fc_b   (ldc = T*V, base = t*V)
    linear8(h_bf, 768, fcw_bf, 768, out + (size_t)t * 32000, (long)16 * 32000,
            fc_b, 32, 32000, 768, 0);
  }

  // hT, cT appended after logits (16384000 floats)
  copy_state_kernel<<<(49152 + 255) / 256, 256, 0, stream>>>(h_f, c_f, out + 16384000);
}
